// Attention_40020505264416
// MI455X (gfx1250) — compile-verified, matrix operand order checked by reading, not run on hardware
//
#include <hip/hip_runtime.h>
#include <hip/hip_bf16.h>

// MI455X / gfx1250 implementation.
// Attention + MLP, all matmuls via V_WMMA_F32_16X16X32_BF16 (wave32),
// MLP GEMMs stage tiles with GLOBAL_LOAD_ASYNC_TO_LDS_B128 (ASYNCcnt DMA).
//
// Shapes: B=4, H=16, N=1024, C=64, D=H*C=1024.
// ctx = (softmax(q k^T / 8) + pe) @ v   -> [B,N,D] in workspace
// h1  = silu(ctx @ w1 + b1)             -> workspace
// out = h1 @ w2 + b2                    -> d_out

typedef __attribute__((ext_vector_type(16))) __bf16 v16bf;
typedef __attribute__((ext_vector_type(8)))  float  v8f;

// 16-bit A/B fragment K-index for element i (0..15) given lane-half (0/1).
// From ISA 7.12.2: VGPR0..3 hold K={0..7}+8*half, VGPR4..7 hold K={16..23}+8*half.
__device__ __forceinline__ int kmap16(int i, int hi) {
    return ((i < 8) ? i : (i + 8)) + (hi << 3);
}

__device__ __forceinline__ v8f wmma_bf16(v16bf a, v16bf b, v8f c) {
    return __builtin_amdgcn_wmma_f32_16x16x32_bf16(
        /*neg_a=*/false, a, /*neg_b=*/false, b,
        /*c_mod=*/(short)0, c, /*reuse_a=*/false, /*reuse_b=*/false);
}

// Generic pointer to LDS: low 32 bits are the LDS byte address
// (LDS aperture, ISA 10.2).
__device__ __forceinline__ unsigned lds_addr_of(const void* p) {
    return (unsigned)(uintptr_t)p;
}

// GLOBAL_LOAD_ASYNC_TO_LDS_B128, GVS mode: mem = SGPR_base + VGPR_byteoff.
// LDS dest address comes from the VDST VGPR. Tracked by ASYNCcnt.
__device__ __forceinline__ void async_copy_b128(unsigned lds_byte_addr,
                                                const void* sbase,
                                                unsigned byte_off) {
    asm volatile("global_load_async_to_lds_b128 %0, %1, %2"
                 :: "v"(lds_byte_addr), "v"(byte_off), "s"(sbase)
                 : "memory");
}

__device__ __forceinline__ void wait_async0() {
    asm volatile("s_wait_asynccnt 0" ::: "memory");
}

// ---------------------------------------------------------------------------
// Attention: one wave per (b, h, 16-row q tile). grid = B*H*(N/16) = 4096.
// ---------------------------------------------------------------------------
__global__ __launch_bounds__(32)
void attn_kernel(const float* __restrict__ q, const float* __restrict__ k,
                 const float* __restrict__ v, const float* __restrict__ pe,
                 float* __restrict__ xout)
{
    __shared__ __align__(16) float s_sc[16 * 1024];   // 64 KB score/prob tile

    const int lane = threadIdx.x & 31;
    const int hi   = lane >> 4;     // which 16-lane half
    const int ln16 = lane & 15;     // M (A rows) or N (B/C cols) index

    const int tile = blockIdx.x & 63;          // N/16 tiles
    const int bh   = blockIdx.x >> 6;
    const int h    = bh & 15;
    const int b    = bh >> 4;
    const int row0 = tile * 16;

    const size_t headoff = (size_t)(b * 16 + h) * 1024 * 64;
    const float* __restrict__ qb = q + headoff;
    const float* __restrict__ kb = k + headoff;
    const float* __restrict__ vb = v + headoff;
    const float* __restrict__ pb = pe + (size_t)h * 1024 * 1024;

    // Q fragments: A is 16x64, row m = ln16 (same rows for both halves).
    v16bf a0, a1;
#pragma unroll
    for (int i = 0; i < 16; ++i) {
        const int c = kmap16(i, hi);
        a0[i] = (__bf16)qb[(size_t)(row0 + ln16) * 64 + c];
        a1[i] = (__bf16)qb[(size_t)(row0 + ln16) * 64 + c + 32];
    }

    // ---- Pass 1: scores = (q k^T) * C^-0.5, stored to LDS ----
    for (int j = 0; j < 64; ++j) {
        // B = k^T (64 x 16keys): B[c][n] = k[j*16+n][c], n = ln16.
        v16bf kb0, kb1;
#pragma unroll
        for (int i = 0; i < 16; ++i) {
            const int c = kmap16(i, hi);
            kb0[i] = (__bf16)kb[(size_t)(j * 16 + ln16) * 64 + c];
            kb1[i] = (__bf16)kb[(size_t)(j * 16 + ln16) * 64 + c + 32];
        }
        v8f acc = {};
        acc = wmma_bf16(a0, kb0, acc);
        acc = wmma_bf16(a1, kb1, acc);
#pragma unroll
        for (int r = 0; r < 8; ++r)
            s_sc[(r + 8 * hi) * 1024 + j * 16 + ln16] = acc[r] * 0.125f;
    }
    __syncthreads();

    // ---- Softmax in LDS: lane handles row=ln16, cols [hi*512, hi*512+512) ----
    float* srow = &s_sc[ln16 * 1024 + hi * 512];
    float mloc = -1e30f;
    for (int t = 0; t < 512; t += 4) {
        const float4 f = *(const float4*)&srow[t];
        mloc = fmaxf(mloc, fmaxf(fmaxf(f.x, f.y), fmaxf(f.z, f.w)));
    }
    mloc = fmaxf(mloc, __shfl_xor(mloc, 16, 32));   // combine the two halves
    float ssum = 0.0f;
    for (int t = 0; t < 512; t += 4) {
        float4 f = *(float4*)&srow[t];
        f.x = __expf(f.x - mloc); f.y = __expf(f.y - mloc);
        f.z = __expf(f.z - mloc); f.w = __expf(f.w - mloc);
        ssum += (f.x + f.y) + (f.z + f.w);
        *(float4*)&srow[t] = f;
    }
    ssum += __shfl_xor(ssum, 16, 32);
    const float inv = 1.0f / ssum;                  // lane's own row == ln16
    __syncthreads();

    // ---- Pass 2: ctx = (softmax + pe) @ v ----
    v8f o0 = {}, o1 = {}, o2 = {}, o3 = {};
    for (int jp = 0; jp < 32; ++jp) {               // K = 32 keys per step
        const float* __restrict__ perow = pb + (size_t)(row0 + ln16) * 1024 + jp * 32;
        const float* __restrict__ prow  = &s_sc[ln16 * 1024 + jp * 32];
        v16bf pa;
#pragma unroll
        for (int i = 0; i < 16; ++i) {
            const int kk = kmap16(i, hi);
            pa[i] = (__bf16)(prow[kk] * inv + perow[kk]);
        }
        const float* __restrict__ vt = vb + (size_t)(jp * 32) * 64;
        v16bf vf;
#pragma unroll
        for (int i = 0; i < 16; ++i) { const int kk = kmap16(i, hi); vf[i] = (__bf16)vt[(size_t)kk * 64 +  0 + ln16]; }
        o0 = wmma_bf16(pa, vf, o0);
#pragma unroll
        for (int i = 0; i < 16; ++i) { const int kk = kmap16(i, hi); vf[i] = (__bf16)vt[(size_t)kk * 64 + 16 + ln16]; }
        o1 = wmma_bf16(pa, vf, o1);
#pragma unroll
        for (int i = 0; i < 16; ++i) { const int kk = kmap16(i, hi); vf[i] = (__bf16)vt[(size_t)kk * 64 + 32 + ln16]; }
        o2 = wmma_bf16(pa, vf, o2);
#pragma unroll
        for (int i = 0; i < 16; ++i) { const int kk = kmap16(i, hi); vf[i] = (__bf16)vt[(size_t)kk * 64 + 48 + ln16]; }
        o3 = wmma_bf16(pa, vf, o3);
    }

    // Store ctx as x[b*N + n][h*64 + c]  (the [B,N,D] MLP input layout)
#pragma unroll
    for (int r = 0; r < 8; ++r) {
        const size_t rowg = (size_t)(b * 1024 + row0 + r + 8 * hi) * 1024 + h * 64;
        xout[rowg +  0 + ln16] = o0[r];
        xout[rowg + 16 + ln16] = o1[r];
        xout[rowg + 32 + ln16] = o2[r];
        xout[rowg + 48 + ln16] = o3[r];
    }
}

// ---------------------------------------------------------------------------
// GEMM: Y[4096,1024] = act(X[4096,1024] @ W[1024,1024] + bias)
// 256 threads (8 waves) per block; block computes a 128x128 tile.
// Each wave: 16 rows x 128 cols (8 f32 accumulators).
// K staged 32-wide into LDS with async global->LDS B128 DMA.
// grid = (4096/128)*(1024/128) = 32*8 = 256 blocks.
// ---------------------------------------------------------------------------
__global__ __launch_bounds__(256)
void gemm_kernel(const float* __restrict__ X, const float* __restrict__ W,
                 const float* __restrict__ bias, float* __restrict__ Y,
                 int apply_silu)
{
    __shared__ __align__(16) float s_x[128 * 32];   // X tile (rows x K)  16 KB
    __shared__ __align__(16) float s_w[32 * 128];   // W tile (K x cols)  16 KB

    const int tid  = threadIdx.x;
    const int lane = tid & 31;
    const int wid  = tid >> 5;                     // wave id 0..7
    const int hi   = lane >> 4;
    const int ln16 = lane & 15;
    const int tileN = (blockIdx.x & 7)  * 128;
    const int tileM = (blockIdx.x >> 3) * 128;
    const size_t D = 1024;

    const unsigned lds_x = lds_addr_of(&s_x[0]);
    const unsigned lds_w = lds_addr_of(&s_w[0]);

    v8f acc[8] = {};
    for (int ks = 0; ks < 32; ++ks) {              // K steps of 32
        // Async-stage both tiles: 4096 floats each = 4 B128 per thread each.
#pragma unroll
        for (int i = 0; i < 4; ++i) {
            const int e  = (i * 256 + tid) * 4;    // float index 0..4092, 16B-aligned
            const int xr = e >> 5, xc = e & 31;    // X tile coords
            async_copy_b128(lds_x + (unsigned)e * 4, X,
                            (unsigned)(((tileM + xr) * 1024 + ks * 32 + xc) * 4));
            const int wr = e >> 7, wn = e & 127;   // W tile coords
            async_copy_b128(lds_w + (unsigned)e * 4, W,
                            (unsigned)(((ks * 32 + wr) * 1024 + tileN + wn) * 4));
        }
        wait_async0();
        __syncthreads();

        v16bf af;
#pragma unroll
        for (int i = 0; i < 16; ++i)
            af[i] = (__bf16)s_x[(wid * 16 + ln16) * 32 + kmap16(i, hi)];
#pragma unroll
        for (int ct = 0; ct < 8; ++ct) {
            v16bf bf;
#pragma unroll
            for (int i = 0; i < 16; ++i)
                bf[i] = (__bf16)s_w[kmap16(i, hi) * 128 + ct * 16 + ln16];
            acc[ct] = wmma_bf16(af, bf, acc[ct]);
        }
        __syncthreads();                           // before next stage overwrites
    }

#pragma unroll
    for (int ct = 0; ct < 8; ++ct) {
        const int colg = tileN + ct * 16 + ln16;
        const float bv = bias[colg];
#pragma unroll
        for (int r = 0; r < 8; ++r) {
            float yv = acc[ct][r] + bv;
            if (apply_silu) yv = yv * (1.0f / (1.0f + __expf(-yv)));   // SiLU
            Y[(size_t)(tileM + wid * 16 + r + 8 * hi) * D + colg] = yv;
        }
    }
}

// ---------------------------------------------------------------------------
extern "C" void kernel_launch(void* const* d_in, const int* in_sizes, int n_in,
                              void* d_out, int out_size, void* d_ws, size_t ws_size,
                              hipStream_t stream)
{
    (void)in_sizes; (void)n_in; (void)out_size; (void)ws_size;
    const float* q  = (const float*)d_in[0];
    const float* k  = (const float*)d_in[1];
    const float* v  = (const float*)d_in[2];
    const float* pe = (const float*)d_in[3];
    const float* w1 = (const float*)d_in[4];
    const float* b1 = (const float*)d_in[5];
    const float* w2 = (const float*)d_in[6];
    const float* b2 = (const float*)d_in[7];

    float* x  = (float*)d_ws;                         // [4096,1024] ctx, 16 MB
    float* h1 = x + (size_t)4096 * 1024;              // [4096,1024] hidden, 16 MB
    float* out = (float*)d_out;

    attn_kernel<<<4 * 16 * 64, 32, 0, stream>>>(q, k, v, pe, x);
    gemm_kernel<<<256, 256, 0, stream>>>(x,  w1, b1, h1,  1);
    gemm_kernel<<<256, 256, 0, stream>>>(h1, w2, b2, out, 0);
}
